// e3interaction_64931315581500
// MI455X (gfx1250) — matrix operand order repaired and checked by reading, not compiled, not run on hardware
//
#include <hip/hip_runtime.h>
#include <hip/hip_bf16.h>
#include <math.h>

typedef __attribute__((ext_vector_type(16))) _Float16 v16h;
typedef __attribute__((ext_vector_type(8)))  float    v8f;

#define SN 16
#define VN 8
#define SO 8
#define VO 4
#define HDIM 128
#define NBASIS 50
#define NUMEL1 576
#define NUMEL2 3584

#define INV_SQ3 0.57735026918962576f
#define C0A     0.20412414523193154f   /* 1/sqrt(24) */
#define C1A3    0.20412414523193154f   /* sqrt(3/24)/sqrt(3) = 1/sqrt(24) */
#define C0B     0.05590169943749474f   /* 1/sqrt(320) */
#define C1B3    0.0625f                /* sqrt(3/256)/sqrt(3) = 1/16 */

__device__ __forceinline__ v8f wmma_f16(v16h a, v16h b, v8f c) {
    return __builtin_amdgcn_wmma_f32_16x16x32_f16(
        /*neg_a=*/false, a, /*neg_b=*/false, b,
        /*c_mod=*/(short)0, c, /*reuse_a=*/false, /*reuse_b=*/false);
}

// ---------------------------------------------------------------------------
// Pack w2 (H x numel, f32, row-major) into WMMA B-matrix tile layout (f16).
// B tile (32x16 f16) as packed here (contiguous k): element (k,n): lane = n + 16*(k>>4), slot = k&15.
// Storage: [nc][kc][lane][slot] halves, nc = numel/16 tiles, kc = 4 K-chunks.
// ---------------------------------------------------------------------------
__global__ void __launch_bounds__(256) pack_w2_kernel(
    const float* __restrict__ w2, _Float16* __restrict__ outp,
    int numel, int total, float scale)
{
    int idx = blockIdx.x * 256 + threadIdx.x;
    if (idx >= total) return;
    int nc   = idx >> 11;
    int kc   = (idx >> 9) & 3;
    int lane = (idx >> 4) & 31;
    int s    = idx & 15;
    int k = kc * 32 + ((lane >> 4) << 4) + s;
    int n = (nc << 4) + (lane & 15);
    outp[idx] = (_Float16)(w2[(size_t)k * numel + n] * scale);
}

// ---------------------------------------------------------------------------
// Per-edge geometry + radial basis + both 128-wide silu hiddens.
// h1/h2 stored as f16 in WMMA A-matrix layout, 16-edge tiles x 4 K-chunks.
// A (16x32 f16): (m,k): lane = m + 16*(k>>4), slot = k&15 (matches B packing)
// ---------------------------------------------------------------------------
__global__ void __launch_bounds__(256) geom_hidden_kernel(
    const int*   __restrict__ ei,  const float* __restrict__ pos,
    const float* __restrict__ w1l, const float* __restrict__ w1r,
    _Float16* __restrict__ h1p, _Float16* __restrict__ h2p,
    float* __restrict__ shv, int E, float actc, float cdemb)
{
    __shared__ float wl[NBASIS * HDIM];
    __shared__ float wr[NBASIS * HDIM];
    const int tid = threadIdx.x;
    const float inv50 = 0.14142135623730950f;  // 1/sqrt(50)
    for (int i = tid; i < NBASIS * HDIM; i += 256) {
        wl[i] = w1l[i] * inv50;
        wr[i] = w1r[i] * inv50;
    }
    __syncthreads();

    int e = blockIdx.x * 256 + tid;
    if (e >= E) return;
    int rc = ei[e], lg = ei[E + e];
    float vx = pos[(size_t)lg * 3 + 0] - pos[(size_t)rc * 3 + 0];
    float vy = pos[(size_t)lg * 3 + 1] - pos[(size_t)rc * 3 + 1];
    float vz = pos[(size_t)lg * 3 + 2] - pos[(size_t)rc * 3 + 2];
    float d = sqrtf(vx * vx + vy * vy + vz * vz);
    float invd = 1.7320508075688772f / fmaxf(d, 1e-12f);
    shv[(size_t)e * 3 + 0] = vx * invd;
    shv[(size_t)e * 3 + 1] = vy * invd;
    shv[(size_t)e * 3 + 2] = vz * invd;

    float demb[NBASIS];
    const float stepinv = 51.0f / 5.0f;
    for (int i = 0; i < NBASIS; ++i) {
        float diff = (d - (5.0f * (float)(i + 1)) / 51.0f) * stepinv;
        float t0 = diff + 1.0f, t1 = 1.0f - diff;
        float s0 = (t0 > 0.0f) ? expf(-1.0f / t0) : 0.0f;
        float s1 = (t1 > 0.0f) ? expf(-1.0f / t1) : 0.0f;
        demb[i] = cdemb * s0 * s1;
    }

    int t = e >> 4, m = e & 15;
    for (int hh = 0; hh < HDIM; ++hh) {
        float a1 = 0.0f, a2 = 0.0f;
        for (int k = 0; k < NBASIS; ++k) {
            float dv = demb[k];
            a1 += dv * wl[k * HDIM + hh];
            a2 += dv * wr[k * HDIM + hh];
        }
        float h1 = actc * a1 / (1.0f + expf(-a1));
        float h2 = actc * a2 / (1.0f + expf(-a2));
        int kc = hh >> 5, kk = hh & 31;
        int lane = m + ((kk >> 4) << 4);
        int s    = kk & 15;
        size_t idx = (size_t)t * 2048 + (size_t)kc * 512 + (size_t)lane * 16 + s;
        h1p[idx] = (_Float16)h1;
        h2p[idx] = (_Float16)h2;
    }
}

// ---------------------------------------------------------------------------
// Stage A: W1 = h1_tile(16x128) @ lig_w2(128x576) via WMMA into LDS,
// fused epilogue -> s_a(16), v_a(8x3) per edge.
// ---------------------------------------------------------------------------
__global__ void __launch_bounds__(128) stageA_kernel(
    const int* __restrict__ ei, const float* __restrict__ x,
    const _Float16* __restrict__ h1p, const _Float16* __restrict__ wLp,
    const float* __restrict__ shv, float* __restrict__ saVa, int E)
{
    __shared__ float Wl[16 * NUMEL1];
    __shared__ float s1s[16][16];
    __shared__ float v1s[16][24];
    __shared__ float shs[16][3];
    __shared__ float vdot[16][8];

    const int tid  = threadIdx.x;
    const int lane = tid & 31;
    const int wave = tid >> 5;
    const int tile = blockIdx.x;
    const int e0   = tile * 16;

    // gather x[lig] (16 + 24 floats/edge) and sh
    for (int o = tid; o < 16 * 40; o += 128) {
        int i = o / 40, c = o % 40;
        int e = e0 + i;
        float val = 0.0f;
        if (e < E) { int lg = ei[E + e]; val = x[(size_t)lg * 40 + c]; }
        if (c < 16) s1s[i][c] = val; else v1s[i][c - 16] = val;
    }
    if (tid < 48) {
        int i = tid / 3, k = tid % 3;
        int e = e0 + i;
        shs[i][k] = (e < E) ? shv[(size_t)e * 3 + k] : 0.0f;
    }

    // GEMM: A tile reused across all 36 N-chunks; 9 chunks per wave
    const _Float16* Ab = h1p + (size_t)tile * 2048;
    v16h a0 = *(const v16h*)(Ab + (0 * 32 + lane) * 16);
    v16h a1 = *(const v16h*)(Ab + (1 * 32 + lane) * 16);
    v16h a2 = *(const v16h*)(Ab + (2 * 32 + lane) * 16);
    v16h a3 = *(const v16h*)(Ab + (3 * 32 + lane) * 16);
    const int nn = lane & 15;
    const int mg = (lane >> 4) * 8;
    for (int ii = 0; ii < 9; ++ii) {
        int nc = wave * 9 + ii;
        const _Float16* Bb = wLp + (size_t)nc * 2048;
        v8f acc = {};
        acc = wmma_f16(a0, *(const v16h*)(Bb + (0 * 32 + lane) * 16), acc);
        acc = wmma_f16(a1, *(const v16h*)(Bb + (1 * 32 + lane) * 16), acc);
        acc = wmma_f16(a2, *(const v16h*)(Bb + (2 * 32 + lane) * 16), acc);
        acc = wmma_f16(a3, *(const v16h*)(Bb + (3 * 32 + lane) * 16), acc);
#pragma unroll
        for (int g = 0; g < 8; ++g)
            Wl[(mg + g) * NUMEL1 + nc * 16 + nn] = acc[g];
    }
    __syncthreads();

    {   // v1 . sh per (edge,u)
        int i = tid >> 3, u = tid & 7;
        vdot[i][u] = v1s[i][u * 3 + 0] * shs[i][0]
                   + v1s[i][u * 3 + 1] * shs[i][1]
                   + v1s[i][u * 3 + 2] * shs[i][2];
    }
    __syncthreads();

    // epilogue: 640 outputs (16 edges x (16 s_a + 24 v_a))
    for (int o = tid; o < 640; o += 128) {
        int i = o / 40, slot = o % 40;
        int e = e0 + i;
        if (e >= E) continue;
        const float* Wr = &Wl[i * NUMEL1];
        if (slot < 16) {
            int w = slot;
            float acc0 = 0.0f, acc1 = 0.0f;
            for (int u = 0; u < 16; ++u) acc0 += Wr[u * 16 + w] * s1s[i][u];
            for (int u = 0; u < 8;  ++u) acc1 += Wr[448 + u * 16 + w] * vdot[i][u];
            saVa[(size_t)e * 40 + w] = C0A * (acc0 + acc1 * INV_SQ3);
        } else {
            int vv = slot - 16; int w = vv / 3, k = vv % 3;
            float acc0 = 0.0f, acc1 = 0.0f;
            for (int u = 0; u < 16; ++u) acc0 += Wr[256 + u * 8 + w] * s1s[i][u];
            for (int u = 0; u < 8;  ++u) acc1 += Wr[384 + u * 8 + w] * v1s[i][u * 3 + k];
            saVa[(size_t)e * 40 + 16 + vv] = C1A3 * (acc0 * shs[i][k] + acc1);
        }
    }
}

// ---------------------------------------------------------------------------
// Stage B: W2 = h2_tile(16x128) @ rec_w2(128x3584), 7 chunks of 512 columns
// (each chunk is one tensor block type), fused contraction -> out(20)/edge.
// ---------------------------------------------------------------------------
__global__ void __launch_bounds__(128) stageB_kernel(
    const int* __restrict__ ei, const float* __restrict__ x,
    const _Float16* __restrict__ h2p, const _Float16* __restrict__ wRp,
    const float* __restrict__ saVa, float* __restrict__ out, int E)
{
    __shared__ float Wc[16 * 512];
    __shared__ float sas[16][16], vas[16][24], sbs[16][16], vbs[16][24];
    __shared__ float dab[16][64];
    __shared__ float sAcc[16][8];
    __shared__ float vAcc[16][12];

    const int tid  = threadIdx.x;
    const int lane = tid & 31;
    const int wave = tid >> 5;
    const int tile = blockIdx.x;
    const int e0   = tile * 16;

    // gathers: s_a/v_a from ws, s_b/v_b from x[rec]
    for (int o = tid; o < 16 * 40; o += 128) {
        int i = o / 40, c = o % 40;
        int e = e0 + i;
        float va = 0.0f, vb = 0.0f;
        if (e < E) {
            va = saVa[(size_t)e * 40 + c];
            int rc = ei[e];
            vb = x[(size_t)rc * 40 + c];
        }
        if (c < 16) { sas[i][c] = va; sbs[i][c] = vb; }
        else        { vas[i][c - 16] = va; vbs[i][c - 16] = vb; }
    }
    ((float*)sAcc)[tid] = 0.0f;                       // 128 entries
    for (int o = tid; o < 192; o += 128) ((float*)vAcc)[o] = 0.0f;
    __syncthreads();

    for (int o = tid; o < 16 * 64; o += 128) {        // dab[u][v] = v_a[u].v_b[v]
        int i = o >> 6, uv = o & 63, u = uv >> 3, v = uv & 7;
        dab[i][uv] = vas[i][u * 3 + 0] * vbs[i][v * 3 + 0]
                   + vas[i][u * 3 + 1] * vbs[i][v * 3 + 1]
                   + vas[i][u * 3 + 2] * vbs[i][v * 3 + 2];
    }
    __syncthreads();

    const _Float16* Ab = h2p + (size_t)tile * 2048;
    v16h a0 = *(const v16h*)(Ab + (0 * 32 + lane) * 16);
    v16h a1 = *(const v16h*)(Ab + (1 * 32 + lane) * 16);
    v16h a2 = *(const v16h*)(Ab + (2 * 32 + lane) * 16);
    v16h a3 = *(const v16h*)(Ab + (3 * 32 + lane) * 16);
    const int nn = lane & 15;
    const int mg = (lane >> 4) * 8;

    for (int q = 0; q < 7; ++q) {
        // GEMM this 512-column chunk into LDS (32 N-chunks, 8 per wave)
        for (int ii = 0; ii < 8; ++ii) {
            int ncl = wave * 8 + ii;
            int nc  = q * 32 + ncl;
            const _Float16* Bb = wRp + (size_t)nc * 2048;
            v8f acc = {};
            acc = wmma_f16(a0, *(const v16h*)(Bb + (0 * 32 + lane) * 16), acc);
            acc = wmma_f16(a1, *(const v16h*)(Bb + (1 * 32 + lane) * 16), acc);
            acc = wmma_f16(a2, *(const v16h*)(Bb + (2 * 32 + lane) * 16), acc);
            acc = wmma_f16(a3, *(const v16h*)(Bb + (3 * 32 + lane) * 16), acc);
#pragma unroll
            for (int g = 0; g < 8; ++g)
                Wc[(mg + g) * 512 + ncl * 16 + nn] = acc[g];
        }
        __syncthreads();

        if (q < 4) {                 // W000: j = u*128 + v*8 + w, u = 4q..4q+3
            int i = tid >> 3, w = tid & 7;
            const float* Wr = &Wc[i * 512];
            float acc = 0.0f;
            for (int uu = 0; uu < 4; ++uu) {
                float sau = sas[i][q * 4 + uu];
                const float* Wu = Wr + uu * 128;
                for (int v = 0; v < 16; ++v) acc += Wu[v * 8 + w] * sbs[i][v] * sau;
            }
            sAcc[i][w] += acc;
        } else if (q == 4) {         // W011: j = u*32 + v*4 + w
            for (int o = tid; o < 192; o += 128) {
                int i = o / 12, r = o % 12, w = r / 3, k = r % 3;
                const float* Wr = &Wc[i * 512];
                float acc = 0.0f;
                for (int u = 0; u < 16; ++u) {
                    float sau = sas[i][u];
                    for (int v = 0; v < 8; ++v)
                        acc += Wr[u * 32 + v * 4 + w] * vbs[i][v * 3 + k] * sau;
                }
                vAcc[i][r] += acc;
            }
        } else if (q == 5) {         // W101: j = u*64 + v*4 + w
            for (int o = tid; o < 192; o += 128) {
                int i = o / 12, r = o % 12, w = r / 3, k = r % 3;
                const float* Wr = &Wc[i * 512];
                float acc = 0.0f;
                for (int u = 0; u < 8; ++u) {
                    float vau = vas[i][u * 3 + k];
                    for (int v = 0; v < 16; ++v)
                        acc += Wr[u * 64 + v * 4 + w] * sbs[i][v] * vau;
                }
                vAcc[i][r] += acc;
            }
        } else {                     // W110: j = u*64 + v*8 + w
            int i = tid >> 3, w = tid & 7;
            const float* Wr = &Wc[i * 512];
            float acc = 0.0f;
            for (int u = 0; u < 8; ++u)
                for (int v = 0; v < 8; ++v)
                    acc += Wr[u * 64 + v * 8 + w] * dab[i][u * 8 + v];
            sAcc[i][w] += acc * INV_SQ3;
        }
        __syncthreads();
    }

    for (int o = tid; o < 16 * 20; o += 128) {
        int i = o / 20, c = o % 20;
        int e = e0 + i;
        if (e >= E) continue;
        float r = (c < 8) ? (C0B * sAcc[i][c]) : (C1B3 * vAcc[i][c - 8]);
        out[(size_t)e * 20 + c] = r;
    }
}

// ---------------------------------------------------------------------------
extern "C" void kernel_launch(void* const* d_in, const int* in_sizes, int n_in,
                              void* d_out, int out_size, void* d_ws, size_t ws_size,
                              hipStream_t stream) {
    const int*   ei  = (const int*)d_in[0];
    const float* pos = (const float*)d_in[1];
    const float* x   = (const float*)d_in[2];
    const float* lw1 = (const float*)d_in[3];
    const float* lw2 = (const float*)d_in[4];
    const float* rw1 = (const float*)d_in[5];
    const float* rw2 = (const float*)d_in[6];
    float* out = (float*)d_out;

    const int E  = in_sizes[0] / 2;
    const size_t nT = (size_t)((E + 15) / 16);

    char* ws = (char*)d_ws;
    size_t off = 0;
    auto wsalloc = [&](size_t bytes) {
        off = (off + 255) & ~(size_t)255;
        size_t o = off; off += bytes; return o;
    };
    _Float16* h1p  = (_Float16*)(ws + wsalloc(nT * 2048 * sizeof(_Float16)));
    _Float16* h2p  = (_Float16*)(ws + wsalloc(nT * 2048 * sizeof(_Float16)));
    _Float16* wLp  = (_Float16*)(ws + wsalloc((size_t)36  * 2048 * sizeof(_Float16)));
    _Float16* wRp  = (_Float16*)(ws + wsalloc((size_t)224 * 2048 * sizeof(_Float16)));
    float*    shv  = (float*)(ws + wsalloc((size_t)E * 3 * sizeof(float)));
    float*    saVa = (float*)(ws + wsalloc((size_t)E * 40 * sizeof(float)));

    // ACT_C: same 40001-point quadrature as the reference (host, deterministic)
    double dzz = 20.0 / 40000.0, ssum = 0.0;
    for (int i = 0; i <= 40000; ++i) {
        double z   = -10.0 + dzz * (double)i;
        double phi = exp(-0.5 * z * z) * 0.3989422804014326779;
        double sv  = z / (1.0 + exp(-z));
        ssum += sv * sv * phi;
    }
    float actc  = (float)(1.0 / sqrt(ssum * dzz));
    float cdemb = (float)(1.14136 * exp(2.0) * sqrt(50.0));
    const float invsqH = 0.08838834764831845f;  // 1/sqrt(128)

    {   // pack lig_w2 (36 tiles) and rec_w2 (224 tiles)
        int totL = NUMEL1 * HDIM;   // 73728
        int totR = NUMEL2 * HDIM;   // 458752
        pack_w2_kernel<<<(totL + 255) / 256, 256, 0, stream>>>(lw2, wLp, NUMEL1, totL, invsqH);
        pack_w2_kernel<<<(totR + 255) / 256, 256, 0, stream>>>(rw2, wRp, NUMEL2, totR, invsqH);
    }

    geom_hidden_kernel<<<(E + 255) / 256, 256, 0, stream>>>(
        ei, pos, lw1, rw1, h1p, h2p, shv, E, actc, cdemb);

    stageA_kernel<<<(unsigned)nT, 128, 0, stream>>>(ei, x, h1p, wLp, shv, saVa, E);

    stageB_kernel<<<(unsigned)nT, 128, 0, stream>>>(ei, x, h2p, wRp, saVa, out, E);
}
